// FP8Linear_17128329576982
// MI455X (gfx1250) — 8-bit matrix operand order repaired and checked by reading, compiled, not run on hardware
//
#include <hip/hip_runtime.h>
#include <hip/hip_fp16.h>

constexpr int GM = 4 * 4096;
constexpr int GN = 8192;
constexpr int GK = 2048;

constexpr int BM = 128;
constexpr int BN = 128;
constexpr int BK = 128;
constexpr int LDS_STRIDE = BK + 16;
constexpr int TILE_BYTES = BM * LDS_STRIDE;

typedef __attribute__((ext_vector_type(16))) int   v16i;
typedef __attribute__((ext_vector_type(8)))  float v8f;

__device__ __forceinline__ void async_copy_b128(unsigned lds_off,
                                                const unsigned char* gptr) {
    asm volatile("global_load_async_to_lds_b128 %0, %1, off"
                 :: "v"(lds_off), "v"((unsigned long long)(uintptr_t)gptr)
                 : "memory");
}
__device__ __forceinline__ void wait_async_le8() {
    asm volatile("s_wait_asynccnt 0x8" ::: "memory");
}
__device__ __forceinline__ void wait_async_0() {
    asm volatile("s_wait_asynccnt 0x0" ::: "memory");
}

#if defined(__has_builtin) && __has_builtin(__builtin_amdgcn_cvt_pk_fp8_f32)
#define HAVE_CVT_PK_FP8 1
#else
#define HAVE_CVT_PK_FP8 0
__device__ inline unsigned char sw_f32_to_e4m3(float x) {
    unsigned sign = (__float_as_uint(x) >> 24) & 0x80u;
    float a = fabsf(x);
    if (!(a > 0.0f)) return (unsigned char)sign;
    if (a > 448.0f) a = 448.0f;
    int e; float m = frexpf(a, &e);
    int fe = e - 1;
    if (fe < -6) {
        int q = (int)rintf(ldexpf(a, 9));
        if (q > 7) q = 7;
        return (unsigned char)(sign | q);
    }
    int q = (int)rintf(ldexpf(m, 4));
    if (q == 16) { q = 8; fe++; }
    if (fe > 8) return (unsigned char)(sign | 0x7E);
    return (unsigned char)(sign | ((fe + 7) << 3) | (q & 7));
}
#endif

__global__ void __launch_bounds__(256)
amax_kernel(const float4* __restrict__ src, int n4, unsigned* __restrict__ slot) {
    __shared__ unsigned red[256];
    float m = 0.0f;
    for (int i = blockIdx.x * blockDim.x + threadIdx.x; i < n4;
         i += gridDim.x * blockDim.x) {
        float4 v = src[i];
        m = fmaxf(m, fmaxf(fmaxf(fabsf(v.x), fabsf(v.y)),
                           fmaxf(fabsf(v.z), fabsf(v.w))));
    }
    red[threadIdx.x] = __float_as_uint(m);
    __syncthreads();
    for (int s = 128; s > 0; s >>= 1) {
        if (threadIdx.x < (unsigned)s)
            red[threadIdx.x] = max(red[threadIdx.x], red[threadIdx.x + s]);
        __syncthreads();
    }
    if (threadIdx.x == 0) atomicMax(slot, red[0]);
}

__global__ void __launch_bounds__(256)
quant_kernel(const float4* __restrict__ src, unsigned* __restrict__ dst,
             int n4, const unsigned* __restrict__ amax_bits) {
    int i = blockIdx.x * blockDim.x + threadIdx.x;
    if (i >= n4) return;
    float amax = fmaxf(__uint_as_float(*amax_bits), 1e-12f);
    float s = 448.0f / amax;
    float4 v = src[i];
#if HAVE_CVT_PK_FP8
    int u = __builtin_amdgcn_cvt_pk_fp8_f32(v.x * s, v.y * s, 0, false);
    u     = __builtin_amdgcn_cvt_pk_fp8_f32(v.z * s, v.w * s, u, true);
    dst[i] = (unsigned)u;
#else
    unsigned u = (unsigned)sw_f32_to_e4m3(v.x * s)
               | ((unsigned)sw_f32_to_e4m3(v.y * s) << 8)
               | ((unsigned)sw_f32_to_e4m3(v.z * s) << 16)
               | ((unsigned)sw_f32_to_e4m3(v.w * s) << 24);
    dst[i] = u;
#endif
}

__global__ void __launch_bounds__(256)
fp8_gemm_kernel(const unsigned char* __restrict__ A,
                const unsigned char* __restrict__ B,
                const __half* __restrict__ bias,
                __half* __restrict__ out,
                const unsigned* __restrict__ amax_bits) {
    __shared__ unsigned char sA[2][TILE_BYTES];
    __shared__ unsigned char sB[2][TILE_BYTES];

    const int tid  = threadIdx.x;
    const int lane = tid & 31;
    const int wave = tid >> 5;
    const int m0 = blockIdx.y * BM;
    const int n0 = blockIdx.x * BN;

    const int wm = (wave & 1) * 64;
    const int wn = (wave >> 1) * 32;

    v8f acc[4][2] = {};

    const unsigned ldsA0 = (unsigned)(uintptr_t)(&sA[0][0]);
    const unsigned ldsB0 = (unsigned)(uintptr_t)(&sB[0][0]);

    auto issue_tile = [&](int kt, int buf) {
        const unsigned char* pa = A + (long)m0 * GK + kt * BK;
        const unsigned char* pb = B + (long)n0 * GK + kt * BK;
        const unsigned offA = ldsA0 + buf * TILE_BYTES;
        const unsigned offB = ldsB0 + buf * TILE_BYTES;
#pragma unroll
        for (int i = 0; i < 4; ++i) {
            int c = tid + i * 256;
            int r = c >> 3;
            int col = (c & 7) * 16;
            async_copy_b128(offA + r * LDS_STRIDE + col, pa + (long)r * GK + col);
            async_copy_b128(offB + r * LDS_STRIDE + col, pb + (long)r * GK + col);
        }
    };

    issue_tile(0, 0);

    constexpr int KT = GK / BK;
    const int half = lane >> 4;
    const int l16  = lane & 15;

    for (int kt = 0; kt < KT; ++kt) {
        const int cur = kt & 1;
        if (kt + 1 < KT) {
            __syncthreads();
            issue_tile(kt + 1, cur ^ 1);
            wait_async_le8();
        } else {
            wait_async_0();
        }
        __syncthreads();

        v16i afrag[4];
#pragma unroll
        for (int mi = 0; mi < 4; ++mi) {
            const unsigned char* row = &sA[cur][(wm + mi * 16 + l16) * LDS_STRIDE];
            const int koff = half * 16;
#pragma unroll
            for (int j = 0; j < 8; ++j) {
                unsigned long long d =
                    *(const unsigned long long*)(row + koff + 32 * (j >> 1) + 8 * (j & 1));
                afrag[mi][2 * j]     = (int)(unsigned)(d & 0xffffffffull);
                afrag[mi][2 * j + 1] = (int)(unsigned)(d >> 32);
            }
        }
        v16i bfrag[2];
#pragma unroll
        for (int ni = 0; ni < 2; ++ni) {
            const unsigned char* row = &sB[cur][(wn + ni * 16 + l16) * LDS_STRIDE];
            const int koff = half * 16;
#pragma unroll
            for (int j = 0; j < 4; ++j) {
                uint4 d = *(const uint4*)(row + koff + 32 * j);
                bfrag[ni][4 * j]     = (int)d.x;
                bfrag[ni][4 * j + 1] = (int)d.y;
                bfrag[ni][4 * j + 2] = (int)d.z;
                bfrag[ni][4 * j + 3] = (int)d.w;
            }
        }
#pragma unroll
        for (int mi = 0; mi < 4; ++mi)
#pragma unroll
            for (int ni = 0; ni < 2; ++ni)
                acc[mi][ni] = __builtin_amdgcn_wmma_f32_16x16x128_fp8_fp8(
                    afrag[mi], bfrag[ni], (short)0, acc[mi][ni], false, false);
    }

    const float amx = fmaxf(__uint_as_float(amax_bits[0]), 1e-12f);
    const float amw = fmaxf(__uint_as_float(amax_bits[1]), 1e-12f);
    const float scale = (amx / 448.0f) * (amw / 448.0f);

#pragma unroll
    for (int mi = 0; mi < 4; ++mi) {
#pragma unroll
        for (int ni = 0; ni < 2; ++ni) {
            const int col = n0 + wn + ni * 16 + l16;
            const __half hb = bias[col];
#pragma unroll
            for (int r = 0; r < 8; ++r) {
                const int row = m0 + wm + mi * 16 + r + half * 8;
                __half h = __float2half(acc[mi][ni][r] * scale);
                h = __hadd(h, hb);
                __builtin_nontemporal_store(
                    __half_as_ushort(h),
                    (unsigned short*)(out + (long)row * GN + col));
            }
        }
    }
}

extern "C" void kernel_launch(void* const* d_in, const int* in_sizes, int n_in,
                              void* d_out, int out_size, void* d_ws, size_t ws_size,
                              hipStream_t stream) {
    const float*  x    = (const float*)d_in[0];
    const float*  w    = (const float*)d_in[1];
    const __half* bias = (const __half*)d_in[2];
    __half*       out  = (__half*)d_out;

    unsigned*      amax = (unsigned*)d_ws;
    unsigned char* xq   = (unsigned char*)d_ws + 256;
    unsigned char* wq   = xq + (size_t)GM * GK;

    hipMemsetAsync(d_ws, 0, 256, stream);

    const int nx4 = GM * GK / 4;
    const int nw4 = GN * GK / 4;

    amax_kernel<<<1024, 256, 0, stream>>>((const float4*)x, nx4, amax + 0);
    amax_kernel<<<1024, 256, 0, stream>>>((const float4*)w, nw4, amax + 1);

    quant_kernel<<<(nx4 + 255) / 256, 256, 0, stream>>>(
        (const float4*)x, (unsigned*)xq, nx4, amax + 0);
    quant_kernel<<<(nw4 + 255) / 256, 256, 0, stream>>>(
        (const float4*)w, (unsigned*)wq, nw4, amax + 1);

    dim3 grid(GN / BN, GM / BM);
    fp8_gemm_kernel<<<grid, 256, 0, stream>>>(xq, wq, bias, out, amax);
}
